// RelationSensitiveAttention_76252849373446
// MI455X (gfx1250) — compile-verified, matrix operand order checked by reading, not run on hardware
//
#include <hip/hip_runtime.h>
#include <hip/hip_bf16.h>

#define BATCH 4
#define SEQ   2048          // 2N
#define NHALF 1024          // N
#define FDIM  1024
#define HEADS 16
#define DHEAD 64

typedef __attribute__((ext_vector_type(16))) _Float16 v16h;
typedef __attribute__((ext_vector_type(8)))  float    v8f;

union FragU { v16h v; unsigned int u[8]; };

// Compiler-only fence: stops IR-level reordering of LDS ops without emitting
// barriers or counter drains. Hardware keeps same-wave DS ops in order
// and the compiler inserts s_wait_dscnt
// for the load->WMMA register dependency, so intra-wave LDS store->load
// round-trips need no s_barrier.
__device__ __forceinline__ void compiler_fence() { asm volatile("" ::: "memory"); }

// v_permlane16_b32 wrapper: permutes within each 16-lane row; nibble j of
// (sel1:sel0) = source lane (0-15 in-row) for in-row lane j.
__device__ __forceinline__ float pl16(float v, unsigned s0, unsigned s1) {
  union { float f; int i; } a, r;
  a.f = v;
  r.i = __builtin_amdgcn_permlane16(a.i, a.i, (int)s0, (int)s1, false, false);
  return r.f;
}

// Max-reduce across the 16 lanes of each half-wave row (xor tree on VALU,
// no LDS pipe / DScnt serialization like ds_bpermute-based __shfl_xor).
__device__ __forceinline__ float rowmax16(float v) {
  v = fmaxf(v, pl16(v, 0x67452301u, 0xEFCDAB89u));  // xor 1
  v = fmaxf(v, pl16(v, 0x54761032u, 0xDCFE98BAu));  // xor 2
  v = fmaxf(v, pl16(v, 0x32107654u, 0xBA98FEDCu));  // xor 4
  v = fmaxf(v, pl16(v, 0xFEDCBA98u, 0x76543210u));  // xor 8
  return v;
}

// Load a 16x32 f16 A/B-operand fragment per the CDNA5 WMMA VGPR layout:
// lane L holds row r = L&15; half = L>>4; VGPR j<4 -> K = 2j + 8*half,
// VGPR j>=4 -> K = 16 + 2(j-4) + 8*half. Each VGPR = 2 consecutive f16.
// `base` points at (row 0, k 0) of the fragment; ldHalf = row stride in f16.
__device__ __forceinline__ v16h load_frag16(const _Float16* __restrict__ base, int ldHalf) {
  const int lane = threadIdx.x & 31;
  const int r  = lane & 15;
  const int hf = lane >> 4;
  const unsigned int* row = reinterpret_cast<const unsigned int*>(base + (size_t)r * ldHalf);
  FragU f;
#pragma unroll
  for (int j = 0; j < 4; ++j) f.u[j]     = row[j + 4 * hf];
#pragma unroll
  for (int j = 0; j < 4; ++j) f.u[4 + j] = row[8 + j + 4 * hf];
  return f.v;
}

__device__ __forceinline__ v8f wmma_f16(v16h a, v16h b, v8f c) {
  return __builtin_amdgcn_wmma_f32_16x16x32_f16(false, a, false, b, (short)0, c, false, false);
}

// ---------------------------------------------------------------------------
// x = concat(broadcast(scene), feature_map) converted to f16: [B, SEQ, F]
// ---------------------------------------------------------------------------
__global__ void build_x_kernel(const float* __restrict__ fm,
                               const float* __restrict__ st,
                               _Float16* __restrict__ xh) {
  size_t idx = (size_t)blockIdx.x * blockDim.x + threadIdx.x;
  if (idx >= (size_t)BATCH * SEQ * FDIM) return;
  int f = (int)(idx % FDIM);
  size_t rb = idx / FDIM;
  int r = (int)(rb % SEQ);
  int b = (int)(rb / SEQ);
  float v = (r < NHALF) ? st[(size_t)b * FDIM + f]
                        : fm[((size_t)b * NHALF + (r - NHALF)) * FDIM + f];
  xh[idx] = (_Float16)v;
}

__global__ void cvt_f32_to_f16_kernel(const float* __restrict__ in,
                                      _Float16* __restrict__ out, int n) {
  int idx = blockIdx.x * blockDim.x + threadIdx.x;
  if (idx < n) out[idx] = (_Float16)in[idx];
}

// ---------------------------------------------------------------------------
// GEMM: Out = X @ W^T + bias, X [B, Srows, F] f16, W [F, F] f16 (row-major,
// so W rows are the B-operand columns of X@W^T and load with the same
// fragment layout as A).
// MODE 0: f16 out, head-major [b, h, s, d], value = (acc+bias)*scale
// MODE 1: f16 out, transposed  [b, h, d, s] (for V, PV B-operand layout)
// MODE 2: f32 out, row-major   [row, F] (final projection, grid.z == 1)
// Block: 256 threads = 8 waves, tile 64(M) x 128(N); wave tile 32x32.
// ---------------------------------------------------------------------------
template<int MODE>
__global__ void gemm_xwt_kernel(const _Float16* __restrict__ X,
                                const _Float16* __restrict__ W,
                                const float* __restrict__ bias,
                                void* __restrict__ outp,
                                int Srows, float scale) {
  const int w    = threadIdx.x >> 5;
  const int wm   = w & 1;        // 2 wave rows
  const int wn   = w >> 1;       // 4 wave cols
  const int b    = blockIdx.z;
  const int m0   = blockIdx.x * 64  + wm * 32;
  const int n0   = blockIdx.y * 128 + wn * 32;
  const _Float16* Xb = X + (size_t)b * Srows * FDIM;

  v8f acc[2][2] = {};
  for (int k0 = 0; k0 < FDIM; k0 += 32) {
    v16h a0 = load_frag16(Xb + (size_t)m0        * FDIM + k0, FDIM);
    v16h a1 = load_frag16(Xb + (size_t)(m0 + 16) * FDIM + k0, FDIM);
    v16h b0 = load_frag16(W  + (size_t)n0        * FDIM + k0, FDIM);
    v16h b1 = load_frag16(W  + (size_t)(n0 + 16) * FDIM + k0, FDIM);
    acc[0][0] = wmma_f16(a0, b0, acc[0][0]);
    acc[0][1] = wmma_f16(a0, b1, acc[0][1]);
    acc[1][0] = wmma_f16(a1, b0, acc[1][0]);
    acc[1][1] = wmma_f16(a1, b1, acc[1][1]);
  }

  const int lane = threadIdx.x & 31;
  const int nn = lane & 15, hf = lane >> 4;
#pragma unroll
  for (int tm = 0; tm < 2; ++tm) {
#pragma unroll
    for (int tn = 0; tn < 2; ++tn) {
      const int nCol = n0 + tn * 16 + nn;
      const float bv = bias[nCol];
#pragma unroll
      for (int i = 0; i < 8; ++i) {
        const int mRow = m0 + tm * 16 + i + 8 * hf;
        const float v = (acc[tm][tn][i] + bv) * scale;
        if (MODE == 0) {
          _Float16* out = (_Float16*)outp;
          const int h = nCol >> 6, d = nCol & 63;
          out[(((size_t)b * HEADS + h) * Srows + mRow) * DHEAD + d] = (_Float16)v;
        } else if (MODE == 1) {
          _Float16* out = (_Float16*)outp;
          const int h = nCol >> 6, d = nCol & 63;
          out[(((size_t)b * HEADS + h) * DHEAD + d) * (size_t)Srows + mRow] = (_Float16)v;
        } else {
          float* out = (float*)outp;
          out[((size_t)b * Srows + mRow) * FDIM + nCol] = v;
        }
      }
    }
  }
}

// ---------------------------------------------------------------------------
// Flash attention with additive relation bias.
// Q,K: f16 [b,h,s,d] (Q pre-scaled by 1/sqrt(D)); Vt: f16 [b,h,d,s].
// Each wave owns a 16-row query tile and a private LDS slice; streams 32
// keys per step. Softmax row-max uses v_permlane16 (VALU) and row-sums are
// folded into the matrix pipe as P @ ones, so the loop has no LDS-pipe
// reduction chain and no s_barrier.
// Output: f16 attn [b, s, h*64+d]  (= [B, SEQ, F] row-major for final GEMM).
// ---------------------------------------------------------------------------
__global__ void attention_kernel(const _Float16* __restrict__ Q,
                                 const _Float16* __restrict__ K,
                                 const _Float16* __restrict__ Vt,
                                 const float* __restrict__ rel,
                                 _Float16* __restrict__ attnOut) {
  __shared__ __align__(16) _Float16 pbuf[8][16 * 34];   // per-wave P transpose tile

  const int w    = threadIdx.x >> 5;
  const int lane = threadIdx.x & 31;
  const int nn = lane & 15, hf = lane >> 4;
  const int h = blockIdx.y, b = blockIdx.z;
  const int q0 = (blockIdx.x * 8 + w) * 16;

  const _Float16* Qb = Q  + ((size_t)b * HEADS + h) * SEQ   * DHEAD;
  const _Float16* Kb = K  + ((size_t)b * HEADS + h) * SEQ   * DHEAD;
  const _Float16* Vb = Vt + ((size_t)b * HEADS + h) * DHEAD * SEQ;

  const v16h aq0 = load_frag16(Qb + (size_t)q0 * DHEAD,      DHEAD);
  const v16h aq1 = load_frag16(Qb + (size_t)q0 * DHEAD + 32, DHEAD);

  // B operand of all-ones: every element f16 1.0 -> P @ ones = row sums in
  // every column of the C tile.
  v16h ones;
  {
    FragU f;
#pragma unroll
    for (int j = 0; j < 8; ++j) f.u[j] = 0x3C003C00u;   // {1.0h, 1.0h}
    ones = f.v;
  }

  v8f o[4] = {};
  v8f osum = {};          // running softmax denominators (row sums)
  float mrun[8];
#pragma unroll
  for (int i = 0; i < 8; ++i) mrun[i] = -1e30f;

  _Float16* myp = pbuf[w];
  const float* relBase = rel + (size_t)(q0 + 8 * hf) * SEQ + nn;

  for (int kb = 0; kb < SEQ; kb += 32) {
    compiler_fence();   // keep this iteration's LDS stores below last iter's loads

    // S(16x32) = Qtile(16x64) @ K[kb:kb+32]^T  -> two 16x16 C tiles
    v8f s0 = {}, s1 = {};
    {
      v16h bk;
      bk = load_frag16(Kb + (size_t)kb        * DHEAD,      DHEAD); s0 = wmma_f16(aq0, bk, s0);
      bk = load_frag16(Kb + (size_t)kb        * DHEAD + 32, DHEAD); s0 = wmma_f16(aq1, bk, s0);
      bk = load_frag16(Kb + (size_t)(kb + 16) * DHEAD,      DHEAD); s1 = wmma_f16(aq0, bk, s1);
      bk = load_frag16(Kb + (size_t)(kb + 16) * DHEAD + 32, DHEAD); s1 = wmma_f16(aq1, bk, s1);
    }

    // + relation bias (Q already carries the 1/sqrt(D) scale)
#pragma unroll
    for (int i = 0; i < 8; ++i) {
      s0[i] += relBase[(size_t)i * SEQ + kb];
      s1[i] += relBase[(size_t)i * SEQ + kb + 16];
    }

    // Online softmax; C-layout rows span the 16 lanes of a half-wave.
    // p values go straight to LDS; row sums come later from P @ ones.
#pragma unroll
    for (int i = 0; i < 8; ++i) {
      const float a0 = s0[i], a1 = s1[i];
      const float mx = rowmax16(fmaxf(a0, a1));
      const float mnew = fmaxf(mrun[i], mx);
      const float corr = __expf(mrun[i] - mnew);
      mrun[i] = mnew;
      const float p0 = __expf(a0 - mnew);
      const float p1 = __expf(a1 - mnew);
      o[0][i] *= corr; o[1][i] *= corr; o[2][i] *= corr; o[3][i] *= corr;
      osum[i] *= corr;
      const int r = i + 8 * hf;                 // C-layout -> A-layout transpose
      myp[r * 34 + nn]      = (_Float16)p0;
      myp[r * 34 + 16 + nn] = (_Float16)p1;
    }

    compiler_fence();   // LDS stores above complete (in-order DS) before reload
    const v16h ap = load_frag16(myp, 34);

    // O(16x64) += P(16x32) @ V[kb:kb+32, :]; Vt rows are contiguous in keys.
    // Row sums ride the matrix pipe: osum += P @ ones.
#pragma unroll
    for (int t = 0; t < 4; ++t) {
      v16h bv = load_frag16(Vb + (size_t)(t * 16) * SEQ + kb, SEQ);
      o[t] = wmma_f16(ap, bv, o[t]);
    }
    osum = wmma_f16(ap, ones, osum);
  }

  // Normalize and store attn output as f16 [b, s, h*64 + d].
#pragma unroll
  for (int i = 0; i < 8; ++i) {
    const int mRow = q0 + i + 8 * hf;
    const float inv = 1.0f / osum[i];
#pragma unroll
    for (int t = 0; t < 4; ++t) {
      attnOut[((size_t)b * SEQ + mRow) * FDIM + h * DHEAD + t * 16 + nn] =
          (_Float16)(o[t][i] * inv);
    }
  }
}

// ---------------------------------------------------------------------------
extern "C" void kernel_launch(void* const* d_in, const int* in_sizes, int n_in,
                              void* d_out, int out_size, void* d_ws, size_t ws_size,
                              hipStream_t stream) {
  const float* fm  = (const float*)d_in[0];   // [B, N, F]
  const float* st  = (const float*)d_in[1];   // [B, F]
  const float* rel = (const float*)d_in[2];   // [2N, 2N]
  const float* Wq  = (const float*)d_in[3];
  const float* bq  = (const float*)d_in[4];
  const float* Wk  = (const float*)d_in[5];
  const float* bk  = (const float*)d_in[6];
  const float* Wv  = (const float*)d_in[7];
  const float* bv  = (const float*)d_in[8];
  const float* Wo  = (const float*)d_in[9];
  const float* bo  = (const float*)d_in[10];

  char* ws = (char*)d_ws;
  const size_t MB = 1024 * 1024;
  _Float16* xh    = (_Float16*)(ws + 0);        // 16 MiB : [B, SEQ, F]
  _Float16* wqh   = (_Float16*)(ws + 16 * MB);  //  2 MiB
  _Float16* wkh   = (_Float16*)(ws + 18 * MB);
  _Float16* wvh   = (_Float16*)(ws + 20 * MB);
  _Float16* woh   = (_Float16*)(ws + 22 * MB);
  _Float16* qh    = (_Float16*)(ws + 24 * MB);  // 16 MiB : [B,H,SEQ,D]
  _Float16* kh    = (_Float16*)(ws + 40 * MB);  // 16 MiB
  _Float16* vt    = (_Float16*)(ws + 56 * MB);  // 16 MiB : [B,H,D,SEQ]
  _Float16* attnh = (_Float16*)(ws + 72 * MB);  // 16 MiB : [B,SEQ,F]

  // 1) f16 staging
  {
    const size_t tot = (size_t)BATCH * SEQ * FDIM;
    build_x_kernel<<<dim3((unsigned)((tot + 255) / 256)), 256, 0, stream>>>(fm, st, xh);
    const int nw = FDIM * FDIM;
    cvt_f32_to_f16_kernel<<<dim3(nw / 256), 256, 0, stream>>>(Wq, wqh, nw);
    cvt_f32_to_f16_kernel<<<dim3(nw / 256), 256, 0, stream>>>(Wk, wkh, nw);
    cvt_f32_to_f16_kernel<<<dim3(nw / 256), 256, 0, stream>>>(Wv, wvh, nw);
    cvt_f32_to_f16_kernel<<<dim3(nw / 256), 256, 0, stream>>>(Wo, woh, nw);
  }

  // 2) QKV projections (Q pre-scaled by 1/sqrt(D) = 0.125)
  {
    dim3 grid(SEQ / 64, FDIM / 128, BATCH);
    gemm_xwt_kernel<0><<<grid, 256, 0, stream>>>(xh, wqh, bq, qh, SEQ, 0.125f);
    gemm_xwt_kernel<0><<<grid, 256, 0, stream>>>(xh, wkh, bk, kh, SEQ, 1.0f);
    gemm_xwt_kernel<1><<<grid, 256, 0, stream>>>(xh, wvh, bv, vt, SEQ, 1.0f);
  }

  // 3) relation-biased flash attention
  {
    dim3 grid(SEQ / (16 * 8), HEADS, BATCH);
    attention_kernel<<<grid, 256, 0, stream>>>(qh, kh, vt, rel, attnh);
  }

  // 4) output projection: all B*SEQ rows as one GEMM, f32 out
  {
    dim3 grid((BATCH * SEQ) / 64, FDIM / 128, 1);
    gemm_xwt_kernel<2><<<grid, 256, 0, stream>>>(attnh, woh, bo, d_out,
                                                 BATCH * SEQ, 1.0f);
  }
}
